// VectorQuantizer_29798483099861
// MI455X (gfx1250) — matrix operand order repaired and checked by reading, not compiled, not run on hardware
//
#include <hip/hip_runtime.h>

typedef __attribute__((ext_vector_type(16))) __bf16   v16bf;
typedef __attribute__((ext_vector_type(8)))  float    v8f;
typedef __attribute__((ext_vector_type(4)))  unsigned int u32x4;
typedef __attribute__((ext_vector_type(4)))  float    f32x4;

#define CODE_DIM   256
#define NUM_CODES  4096
#define EMA_DECAY  0.99f
#define ONE_MINUS_DECAY 0.01f
#define EPS_F      1e-5f

// ---------- helpers ----------
__device__ __forceinline__ unsigned short f32_to_bf16_rne(float x) {
    unsigned int u = __float_as_uint(x);
    unsigned int r = u + 0x7FFFu + ((u >> 16) & 1u);
    return (unsigned short)(r >> 16);
}

// gfx1250 async copy: 16B/lane global -> LDS, tracked by ASYNCcnt.
// LDS byte offset = low 32 bits of the generic shared-memory address
__device__ __forceinline__ void async_cp16(void* lds_ptr, const void* gptr) {
    unsigned lds_off = (unsigned)(unsigned long long)lds_ptr;
    asm volatile("global_load_async_to_lds_b128 %0, %1, off"
                 :: "v"(lds_off), "v"(gptr) : "memory");
}

__device__ __forceinline__ void wait_asynccnt0() {
#if __has_builtin(__builtin_amdgcn_s_wait_asynccnt)
    __builtin_amdgcn_s_wait_asynccnt(0);
#else
    asm volatile("s_wait_asynccnt 0x0" ::: "memory");
#endif
}

// ---------- 1) split f32 -> bf16 hi/lo planes ----------
__global__ void split_bf16_kernel(const float* __restrict__ src,
                                  unsigned short* __restrict__ hi,
                                  unsigned short* __restrict__ lo, int n) {
    int i = blockIdx.x * blockDim.x + threadIdx.x;
    if (i < n) {
        float x = src[i];
        unsigned short h = f32_to_bf16_rne(x);
        float hf = __uint_as_float((unsigned int)h << 16);
        hi[i] = h;
        lo[i] = f32_to_bf16_rne(x - hf);
    }
}

// ---------- 2) ||c_k||^2 (one wave per codebook row) ----------
__global__ __launch_bounds__(256) void cnorm_kernel(const float* __restrict__ cb,
                                                    float* __restrict__ cnorm) {
    int wave = threadIdx.x >> 5, lane = threadIdx.x & 31;
    int k = blockIdx.x * 8 + wave;
    const float* r = cb + (size_t)k * CODE_DIM + lane * 8;
    f32x4 a = *(const f32x4*)r;
    f32x4 b = *(const f32x4*)(r + 4);
    float s = a.x*a.x + a.y*a.y + a.z*a.z + a.w*a.w
            + b.x*b.x + b.y*b.y + b.z*b.z + b.w*b.w;
    #pragma unroll
    for (int off = 16; off >= 1; off >>= 1) s += __shfl_xor(s, off, 32);
    if (lane == 0) cnorm[k] = s;
}

// ---------- 3) split-bf16 WMMA GEMM + running argmin ----------
// block = 256 threads = 8 waves; wave w owns rows [blk*128 + w*16, +16).
// Codebook tiles staged once per block into LDS via async copies
// (double-buffered), all 8 waves consume via ds_load_b128.
__global__ __launch_bounds__(256) void gemm_argmin_kernel(
    const unsigned short* __restrict__ z_hi, const unsigned short* __restrict__ z_lo,
    const unsigned short* __restrict__ cb_hi, const unsigned short* __restrict__ cb_lo,
    const float* __restrict__ cnorm, int* __restrict__ idx_out,
    float* __restrict__ idx_f_out)
{
    // [buf][plane][row*256 + d] : 2 * 2 * 8KB = 32KB LDS
    __shared__ unsigned short sB[2][2][16 * CODE_DIM];

    const int tid  = threadIdx.x;
    const int wave = tid >> 5;
    const int lane = tid & 31;
    const int half = lane >> 4;     // K-half selector for A, B fragments
    const int l15  = lane & 15;
    const int m0   = blockIdx.x * 128 + wave * 16;
    const int arow = m0 + l15;      // A-matrix row held by this lane

    // A fragments (16x32 bf16 per chunk): lane holds row=l15; K runs
    // [32c+8*half, +8) and [32c+16+8*half, +8)  -> two b128 loads each.
    v16bf ahi[8], alo[8];
    const unsigned short* zh = z_hi + (size_t)arow * CODE_DIM;
    const unsigned short* zl = z_lo + (size_t)arow * CODE_DIM;
    #pragma unroll
    for (int c = 0; c < 8; ++c) {
        int k0 = 32 * c + 8 * half;
        ((u32x4*)&ahi[c])[0] = *(const u32x4*)(zh + k0);
        ((u32x4*)&ahi[c])[1] = *(const u32x4*)(zh + k0 + 16);
        ((u32x4*)&alo[c])[0] = *(const u32x4*)(zl + k0);
        ((u32x4*)&alo[c])[1] = *(const u32x4*)(zl + k0 + 16);
    }

    float bestv[8];
    int   besti[8];
    #pragma unroll
    for (int r = 0; r < 8; ++r) { bestv[r] = 3.4e38f; besti[r] = 0; }

    const int NT = NUM_CODES / 16;  // 256 N-tiles

    // stage tile 0 into buffer 0: each thread async-copies 4x16B
    {
        const unsigned short* gh = cb_hi + (size_t)0 * CODE_DIM + tid * 8;
        const unsigned short* gl = cb_lo + (size_t)0 * CODE_DIM + tid * 8;
        async_cp16(&sB[0][0][tid * 8],        gh);
        async_cp16(&sB[0][0][tid * 8 + 2048], gh + 2048);
        async_cp16(&sB[0][1][tid * 8],        gl);
        async_cp16(&sB[0][1][tid * 8 + 2048], gl + 2048);
    }

    for (int t = 0; t < NT; ++t) {
        wait_asynccnt0();      // own stage from iteration t-1 complete
        __syncthreads();       // everyone done with compute(t-1) + all stages visible
        const int cur = t & 1;

        if (t + 1 < NT) {      // stage tile t+1 into the other buffer
            const int nxt = (t + 1) & 1;
            const unsigned short* gh = cb_hi + (size_t)(t + 1) * 16 * CODE_DIM + tid * 8;
            const unsigned short* gl = cb_lo + (size_t)(t + 1) * 16 * CODE_DIM + tid * 8;
            async_cp16(&sB[nxt][0][tid * 8],        gh);
            async_cp16(&sB[nxt][0][tid * 8 + 2048], gh + 2048);
            async_cp16(&sB[nxt][1][tid * 8],        gl);
            async_cp16(&sB[nxt][1][tid * 8 + 2048], gl + 2048);
        }

        const int n0  = t * 16;
        const int col = n0 + l15;
        float cn = cnorm[col];

        v8f acc = {};
        // B fragment (32x16): lane holds codebook row l15 of the tile,
        // K in [32c + 8*half, +8) and [32c + 16 + 8*half, +8) -> two ds_load_b128 per plane.
        const unsigned short* ch = &sB[cur][0][l15 * CODE_DIM + 8 * half];
        const unsigned short* cl = &sB[cur][1][l15 * CODE_DIM + 8 * half];
        #pragma unroll
        for (int c = 0; c < 8; ++c) {
            v16bf bhi, blo;
            ((u32x4*)&bhi)[0] = *(const u32x4*)(ch + 32 * c);
            ((u32x4*)&bhi)[1] = *(const u32x4*)(ch + 32 * c + 16);
            ((u32x4*)&blo)[0] = *(const u32x4*)(cl + 32 * c);
            ((u32x4*)&blo)[1] = *(const u32x4*)(cl + 32 * c + 16);
            // split-f32 product: hi*hi + lo*hi + hi*lo   (lo*lo dropped)
            acc = __builtin_amdgcn_wmma_f32_16x16x32_bf16(false, ahi[c], false, bhi,
                                                          (short)0, acc, false, false);
            acc = __builtin_amdgcn_wmma_f32_16x16x32_bf16(false, alo[c], false, bhi,
                                                          (short)0, acc, false, false);
            acc = __builtin_amdgcn_wmma_f32_16x16x32_bf16(false, ahi[c], false, blo,
                                                          (short)0, acc, false, false);
        }

        #pragma unroll
        for (int r = 0; r < 8; ++r) {
            float score = cn - 2.0f * acc[r];   // ||c||^2 - 2 z.c  (||z||^2 const per row)
            if (score < bestv[r] || (score == bestv[r] && col < besti[r])) {
                bestv[r] = score; besti[r] = col;
            }
        }
    }

    // Butterfly argmin across the 16 lanes sharing the same D-row set.
    // D layout: VGPR r -> row M = r + 8*half, col N = l15.
    #pragma unroll
    for (int r = 0; r < 8; ++r) {
        float v = bestv[r];
        int   i = besti[r];
        #pragma unroll
        for (int off = 8; off >= 1; off >>= 1) {
            float ov = __shfl_xor(v, off, 32);
            int   oi = __shfl_xor(i, off, 32);
            if (ov < v || (ov == v && oi < i)) { v = ov; i = oi; }
        }
        if (l15 == 0) {
            int orow = m0 + 8 * half + r;
            idx_out[orow]   = i;
            idx_f_out[orow] = (float)i;
        }
    }
}

// ---------- 4) gather z_q + fused segment-sum scatter (one wave per row) ----------
__global__ __launch_bounds__(256) void gather_scatter_kernel(
    const float* __restrict__ z_e, const float* __restrict__ codebook,
    const int* __restrict__ idx, float* __restrict__ zq_out,
    float* __restrict__ counts, float* __restrict__ emb_ws,
    float* __restrict__ loss_ws)
{
    int wave = threadIdx.x >> 5, lane = threadIdx.x & 31;
    int b = blockIdx.x * 8 + wave;
    int k = idx[b];
    const float* zr = z_e      + (size_t)b * CODE_DIM + lane * 8;
    const float* cr = codebook + (size_t)k * CODE_DIM + lane * 8;
    f32x4 z0 = *(const f32x4*)zr,       z1 = *(const f32x4*)(zr + 4);
    f32x4 c0 = *(const f32x4*)cr,       c1 = *(const f32x4*)(cr + 4);
    float* qo = zq_out + (size_t)b * CODE_DIM + lane * 8;
    *(f32x4*)qo       = c0;
    *(f32x4*)(qo + 4) = c1;

    float d0 = c0.x - z0.x, d1 = c0.y - z0.y, d2 = c0.z - z0.z, d3 = c0.w - z0.w;
    float d4 = c1.x - z1.x, d5 = c1.y - z1.y, d6 = c1.z - z1.z, d7 = c1.w - z1.w;
    float ssq = d0*d0 + d1*d1 + d2*d2 + d3*d3 + d4*d4 + d5*d5 + d6*d6 + d7*d7;
    #pragma unroll
    for (int off = 16; off >= 1; off >>= 1) ssq += __shfl_xor(ssq, off, 32);

    float* eb = emb_ws + (size_t)k * CODE_DIM + lane * 8;
    unsafeAtomicAdd(eb + 0, z0.x); unsafeAtomicAdd(eb + 1, z0.y);
    unsafeAtomicAdd(eb + 2, z0.z); unsafeAtomicAdd(eb + 3, z0.w);
    unsafeAtomicAdd(eb + 4, z1.x); unsafeAtomicAdd(eb + 5, z1.y);
    unsafeAtomicAdd(eb + 6, z1.z); unsafeAtomicAdd(eb + 7, z1.w);
    if (lane == 0) {
        unsafeAtomicAdd(loss_ws, ssq);
        unsafeAtomicAdd(counts + k, 1.0f);
    }
}

// ---------- 5) EMA cluster size + total n + loss ----------
__global__ __launch_bounds__(256) void ema_cs_kernel(
    const float* __restrict__ ema_cs, const float* __restrict__ counts,
    float* __restrict__ out_ncs, float* __restrict__ n_ws,
    float* __restrict__ out_loss, const float* __restrict__ loss_ws, int B)
{
    __shared__ float sdata[256];
    int k = blockIdx.x * 256 + threadIdx.x;
    float ncs = ema_cs[k] * EMA_DECAY + ONE_MINUS_DECAY * counts[k];
    out_ncs[k] = ncs;
    sdata[threadIdx.x] = ncs;
    __syncthreads();
    #pragma unroll
    for (int s = 128; s > 0; s >>= 1) {
        if (threadIdx.x < s) sdata[threadIdx.x] += sdata[threadIdx.x + s];
        __syncthreads();
    }
    if (threadIdx.x == 0) unsafeAtomicAdd(n_ws, sdata[0]);
    if (blockIdx.x == 0 && threadIdx.x == 0)
        *out_loss = 0.25f * (*loss_ws) / ((float)B * (float)CODE_DIM);
}

// ---------- 6) finalize embed_sum + codebook ----------
__global__ __launch_bounds__(256) void finalize_kernel(
    const float* __restrict__ ema_es, const float* __restrict__ emb_ws,
    const float* __restrict__ out_ncs, const float* __restrict__ n_ws,
    float* __restrict__ out_cb, float* __restrict__ out_es)
{
    int k = blockIdx.x, d = threadIdx.x;
    size_t o = (size_t)k * CODE_DIM + d;
    float es = ema_es[o] * EMA_DECAY + ONE_MINUS_DECAY * emb_ws[o];
    out_es[o] = es;
    float n  = *n_ws;
    float cs = (out_ncs[k] + EPS_F) / (n + (float)NUM_CODES * EPS_F) * n;
    out_cb[o] = es / cs;
}

extern "C" void kernel_launch(void* const* d_in, const int* in_sizes, int n_in,
                              void* d_out, int out_size, void* d_ws, size_t ws_size,
                              hipStream_t stream) {
    const float* z_e      = (const float*)d_in[0];
    const float* codebook = (const float*)d_in[1];
    const float* ema_cs   = (const float*)d_in[2];
    const float* ema_es   = (const float*)d_in[3];
    const int B = in_sizes[0] / CODE_DIM;   // 16384
    const int K = in_sizes[1] / CODE_DIM;   // 4096

    // ---- output layout (tuple flattened in return order) ----
    float* out      = (float*)d_out;
    float* out_zq   = out;                              // B*256
    float* out_loss = out_zq + (size_t)B * CODE_DIM;    // 1
    float* out_idx  = out_loss + 1;                     // B (float-encoded)
    float* out_cb   = out_idx + B;                      // K*256
    float* out_ncs  = out_cb + (size_t)K * CODE_DIM;    // K
    float* out_es   = out_ncs + K;                      // K*256

    // ---- workspace layout ----
    char* ws = (char*)d_ws;
    unsigned short* z_hi  = (unsigned short*)ws;                            // B*256*2
    unsigned short* z_lo  = (unsigned short*)(ws + (size_t)B * 512);        // B*256*2
    unsigned short* cb_hi = (unsigned short*)(ws + (size_t)B * 1024);       // K*256*2
    unsigned short* cb_lo = cb_hi + (size_t)K * CODE_DIM;                   // K*256*2
    float* cnorm   = (float*)(cb_lo + (size_t)K * CODE_DIM);                // K
    int*   idx_i   = (int*)(cnorm + K);                                     // B
    float* counts  = (float*)(idx_i + B);       // K   -- zeroed region start
    float* emb_ws  = counts + K;                // K*256
    float* loss_ws = emb_ws + (size_t)K * CODE_DIM;  // 1
    float* n_ws    = loss_ws + 1;                    // 1

    // zero the atomic accumulators (counts|emb_ws|loss|n contiguous)
    size_t zero_bytes = ((size_t)K + (size_t)K * CODE_DIM + 2) * sizeof(float);
    hipMemsetAsync(counts, 0, zero_bytes, stream);

    split_bf16_kernel<<<(B * CODE_DIM + 255) / 256, 256, 0, stream>>>(z_e, z_hi, z_lo, B * CODE_DIM);
    split_bf16_kernel<<<(K * CODE_DIM + 255) / 256, 256, 0, stream>>>(codebook, cb_hi, cb_lo, K * CODE_DIM);
    cnorm_kernel<<<K / 8, 256, 0, stream>>>(codebook, cnorm);
    gemm_argmin_kernel<<<B / 128, 256, 0, stream>>>(z_hi, z_lo, cb_hi, cb_lo, cnorm, idx_i, out_idx);
    gather_scatter_kernel<<<B / 8, 256, 0, stream>>>(z_e, codebook, idx_i, out_zq, counts, emb_ws, loss_ws);
    ema_cs_kernel<<<K / 256, 256, 0, stream>>>(ema_cs, counts, out_ncs, n_ws, out_loss, loss_ws, B);
    finalize_kernel<<<K, 256, 0, stream>>>(ema_es, emb_ws, out_ncs, n_ws, out_cb, out_es);
}
